// w2vCNNText_28424093565669
// MI455X (gfx1250) — matrix operand order repaired and checked by reading, not compiled, not run on hardware
//
#include <hip/hip_runtime.h>

// ---------------------------------------------------------------------------
// Fused TextCNN inference for MI455X (gfx1250, wave32, WMMA).
//   words[512,512] -> gather emb[100000,100] (L2-resident) -> bf16 x in LDS ->
//   unified conv-as-matmul via v_wmma_f32_16x16x32_bf16 (software-pipelined
//   LDS fragment loads) -> relu + per-channel masked max-over-time -> FC[14->5].
// One workgroup per batch row; HBM traffic ~1 MB total.
// ---------------------------------------------------------------------------

typedef __bf16 bf16;
typedef bf16  v16bf __attribute__((ext_vector_type(16)));
typedef float v8f   __attribute__((ext_vector_type(8)));

union Frag { uint4 q[2]; v16bf v; };

// LDS layout (in 32-bit words):
//   x   : [520 rows][68 u32]  (136 bf16 pitch, 128 logical d, rows 512..519 zero)
//   B^T : [16 ch ][336 u32]   (672 bf16 pitch, K = j*128+d, K>=640 zero)
//   red : [8 waves][16 ch] f32
//   feats: [16] f32
#define X_U      (520 * 68)      // 35360
#define B_BASE_U X_U             // 35360
#define B_U      (16 * 336)      // 5376
#define RED_U    (B_BASE_U + B_U)        // 40736
#define FEATS_U  (RED_U + 128)           // 40864
#define TOTAL_U  (FEATS_U + 16)          // 40880  -> 163,520 bytes

__device__ __forceinline__ unsigned f2bf_pk(float lo, float hi) {
  unsigned ul = __float_as_uint(lo);
  unsigned uh = __float_as_uint(hi);
  ul = (ul + 0x7FFFu + ((ul >> 16) & 1u)) >> 16;   // round-to-nearest-even
  uh = (uh + 0x7FFFu + ((uh >> 16) & 1u)) >> 16;
  return (uh << 16) | (ul & 0xFFFFu);
}

__global__ void __launch_bounds__(256, 1)
textcnn_wmma_kernel(const int* __restrict__ words, const float* __restrict__ emb,
                    const float* __restrict__ w2, const float* __restrict__ b2,
                    const float* __restrict__ w3, const float* __restrict__ b3,
                    const float* __restrict__ w4, const float* __restrict__ b4,
                    const float* __restrict__ w5, const float* __restrict__ b5,
                    const float* __restrict__ fcw, const float* __restrict__ fcb,
                    float* __restrict__ out)
{
  extern __shared__ __align__(16) unsigned smemU[];
  unsigned* xU    = smemU;                       // [520][68]
  unsigned* bU    = smemU + B_BASE_U;            // [16][336]
  float*    red   = (float*)(smemU + RED_U);     // [8][16]
  float*    feats = (float*)(smemU + FEATS_U);   // [16]

  const int tid  = threadIdx.x;
  const int wave = tid >> 5;
  const int lane = tid & 31;
  const int lo   = lane & 15;      // row/col index inside 16-wide half
  const int hi   = lane >> 4;      // which K/M half this lane holds
  const int b    = blockIdx.x;

  // ---- Phase 0: build bf16 B^T (weights, K = tap*128 + d) + zero x tail rows
  for (int idx = tid; idx < 16 * 336; idx += 256) {
    int n  = idx / 336;
    int pk = idx - n * 336;
    float v0 = 0.f, v1 = 0.f;
    const float* wsel = nullptr; int ksn = 0, on = 0;
    if      (n < 2)  { wsel = w2; ksn = 2; on = n;     }
    else if (n < 5)  { wsel = w3; ksn = 3; on = n - 2; }
    else if (n < 9)  { wsel = w4; ksn = 4; on = n - 5; }
    else if (n < 14) { wsel = w5; ksn = 5; on = n - 9; }
    const int k0 = pk * 2, k1 = pk * 2 + 1;
    if (wsel) {
      int j0 = k0 >> 7, d0 = k0 & 127;
      int j1 = k1 >> 7, d1 = k1 & 127;
      if (k0 < 640 && d0 < 100 && j0 < ksn) v0 = wsel[(on * 100 + d0) * ksn + j0];
      if (k1 < 640 && d1 < 100 && j1 < ksn) v1 = wsel[(on * 100 + d1) * ksn + j1];
    }
    bU[n * 336 + pk] = f2bf_pk(v0, v1);
  }
  for (int idx = tid; idx < 8 * 68; idx += 256)  // zero tap-overhang rows 512..519
    xU[512 * 68 + idx] = 0u;

  // ---- Phase 1: gather embedding rows, convert fp32 -> bf16, store to LDS
  const int* wrow = words + b * 512;
  for (int r = wave; r < 512; r += 8) {
    const int widx = wrow[r];
    const float2* src = (const float2*)(emb + (size_t)widx * 100);
    unsigned* dst = xU + r * 68;
    if (r + 8 < 512) {                       // pipeline next gather into L2/L0
      const int wnxt = wrow[r + 8];
      __builtin_prefetch(emb + (size_t)wnxt * 100 + lane * 4, 0, 1);
    }
    float2 t0 = src[lane];                   // d = 2*lane, 2*lane+1  (< 64)
    dst[lane] = f2bf_pk(t0.x, t0.y);
    const int pk2 = lane + 32;               // d = 64..127 (zero-pad d >= 100)
    float2 t1 = make_float2(0.f, 0.f);
    if (pk2 < 50) t1 = src[pk2];
    dst[pk2] = f2bf_pk(t1.x, t1.y);
  }
  __syncthreads();

  // ---- Phase 2: conv as WMMA matmul, software-pipelined fragment loads.
  // Bias folded into accumulator init (D = A*B + C with C = bias broadcast).
  const float bv = (lo < 2) ? b2[lo] : (lo < 5) ? b3[lo - 2]
                 : (lo < 9) ? b4[lo - 5] : (lo < 14) ? b5[lo - 9] : 0.f;
  v8f acc[4];
#pragma unroll
  for (int ti = 0; ti < 4; ++ti)
#pragma unroll
    for (int e = 0; e < 8; ++e) acc[ti][e] = bv;

  const int tile0 = wave * 4;                 // 8 waves x 4 tiles = t 0..511
  const unsigned bbase = (unsigned)lo * 336 + (unsigned)hi * 8;

  // Prologue: load fragments for kc = 0 (tap j=0, d-base 0).
  Frag Bf, Af[4];
  Bf.q[0] = *(const uint4*)(bU + bbase);
  Bf.q[1] = *(const uint4*)(bU + bbase + 4);
#pragma unroll
  for (int ti = 0; ti < 4; ++ti) {
    const unsigned p    = (unsigned)((tile0 + ti) * 16 + lo);
    const unsigned aoff = p * 68 + (unsigned)hi * 8;
    Af[ti].q[0] = *(const uint4*)(xU + aoff);
    Af[ti].q[1] = *(const uint4*)(xU + aoff + 4);
  }

  for (int kc = 0; kc < 20; ++kc) {           // K = 640 = 20 chunks of 32
    Frag Bn, An[4];
    if (kc + 1 < 20) {                        // issue next-chunk loads FIRST so
      const int kn = kc + 1;                  // LDS latency overlaps the WMMAs
      const int j  = kn >> 2;                 // conv tap 0..4
      const unsigned dbu  = (unsigned)(kn & 3) * 16;   // d-base within tap (u32)
      const unsigned boff = bbase + (unsigned)kn * 16;
      Bn.q[0] = *(const uint4*)(bU + boff);
      Bn.q[1] = *(const uint4*)(bU + boff + 4);
#pragma unroll
      for (int ti = 0; ti < 4; ++ti) {
        const unsigned p    = (unsigned)((tile0 + ti) * 16 + lo);
        const unsigned aoff = (p + (unsigned)j) * 68 + dbu + (unsigned)hi * 8;
        An[ti].q[0] = *(const uint4*)(xU + aoff);
        An[ti].q[1] = *(const uint4*)(xU + aoff + 4);
      }
    }
#pragma unroll
    for (int ti = 0; ti < 4; ++ti)            // B reused across all 4 M-tiles
      acc[ti] = __builtin_amdgcn_wmma_f32_16x16x32_bf16(
          false, Af[ti].v, false, Bf.v, (short)0, acc[ti], false, false);
    if (kc + 1 < 20) {
      Bf = Bn;
#pragma unroll
      for (int ti = 0; ti < 4; ++ti) Af[ti] = An[ti];
    }
  }

  // ---- Phase 3: relu + per-channel masked max over time
  const int ksz  = (lo < 2) ? 2 : (lo < 5) ? 3 : (lo < 9) ? 4 : (lo < 14) ? 5 : 999;
  const int tmax = 512 - ksz;                // valid t in [0, 512-k]; pad ch -> none
  float m = -3.0e38f;
#pragma unroll
  for (int ti = 0; ti < 4; ++ti)
#pragma unroll
    for (int r = 0; r < 8; ++r) {
      const int pos = (tile0 + ti) * 16 + r + hi * 8;   // C layout: VGPR r = M r / 8+r
      const float v = fmaxf(acc[ti][r], 0.f);
      m = (pos <= tmax) ? fmaxf(m, v) : m;
    }
  m = fmaxf(m, __shfl_xor(m, 16, 32));       // lanes l and l+16 share channel lo
  if (hi == 0) red[wave * 16 + lo] = m;
  __syncthreads();

  if (tid < 16) {
    float f = red[tid];
#pragma unroll
    for (int w = 1; w < 8; ++w) f = fmaxf(f, red[w * 16 + tid]);
    feats[tid] = f;
  }
  __syncthreads();

  // ---- Phase 4: FC 14 -> 5
  if (tid < 5) {
    float a = fcb[tid];
#pragma unroll
    for (int c = 0; c < 14; ++c) a += feats[c] * fcw[tid * 14 + c];
    out[b * 5 + tid] = a;
  }
}

extern "C" void kernel_launch(void* const* d_in, const int* in_sizes, int n_in,
                              void* d_out, int out_size, void* d_ws, size_t ws_size,
                              hipStream_t stream) {
  const int*   words = (const int*)  d_in[0];
  const float* emb   = (const float*)d_in[1];
  const float* w2    = (const float*)d_in[2];
  const float* b2    = (const float*)d_in[3];
  const float* w3    = (const float*)d_in[4];
  const float* b3    = (const float*)d_in[5];
  const float* w4    = (const float*)d_in[6];
  const float* b4    = (const float*)d_in[7];
  const float* w5    = (const float*)d_in[8];
  const float* b5    = (const float*)d_in[9];
  const float* fcw   = (const float*)d_in[10];
  const float* fcb   = (const float*)d_in[11];
  float* out = (float*)d_out;

  const size_t smem = (size_t)TOTAL_U * sizeof(unsigned);  // 163,520 B (<320 KB WGP LDS)
  (void)hipFuncSetAttribute((const void*)textcnn_wmma_kernel,
                            hipFuncAttributeMaxDynamicSharedMemorySize, (int)smem);
  textcnn_wmma_kernel<<<dim3(512), dim3(256), smem, stream>>>(
      words, emb, w2, b2, w3, b3, w4, b4, w5, b5, fcw, fcb, out);
}
